// MixedMoEProjectionLayer_31155692765500
// MI455X (gfx1250) — compile-verified, matrix operand order checked by reading, not run on hardware
//
#include <hip/hip_runtime.h>
#include <hip/hip_bf16.h>

// ---------------------------------------------------------------------------
// MixedMoEProjectionLayer for gfx1250 (MI455X).
// Dense all-expert MoE. bf16 WMMA GEMM (v_wmma_f32_16x16x32_bf16) with:
//   - double-buffered LDS staging via GLOBAL_LOAD_ASYNC_TO_LDS_B128 (ASYNCcnt)
//   - 80-byte LDS row pitch (20-bank stride -> conflict-free 16-row frag reads)
//   - 64x64 wave tiles, 128x256 block tiles, fused bias+activation epilogue
// ---------------------------------------------------------------------------

#define BATCH      8192
#define INPUT_DIM  1024
#define NUM_EXP    4

typedef __attribute__((ext_vector_type(16))) __bf16 bf16x16;
typedef __attribute__((ext_vector_type(8)))  __bf16 bf16x8;
typedef __attribute__((ext_vector_type(8)))  float  f32x8;

// ---------------------------------------------------------------- gate -----
__global__ __launch_bounds__(256)
void gate_kernel(const float* __restrict__ x, const float* __restrict__ gw,
                 const float* __restrict__ gb, float* __restrict__ scores)
{
    int gtid = blockIdx.x * blockDim.x + threadIdx.x;
    int row  = gtid >> 5;
    int lane = threadIdx.x & 31;
    if (row >= BATCH) return;
    const float* xr = x + (size_t)row * INPUT_DIM;
    float s0 = 0.f, s1 = 0.f, s2 = 0.f, s3 = 0.f;
    for (int k = lane; k < INPUT_DIM; k += 32) {
        float xv = xr[k];
        const float* g = gw + k * NUM_EXP;
        s0 += xv * g[0]; s1 += xv * g[1]; s2 += xv * g[2]; s3 += xv * g[3];
    }
    for (int off = 16; off > 0; off >>= 1) {
        s0 += __shfl_xor(s0, off, 32);
        s1 += __shfl_xor(s1, off, 32);
        s2 += __shfl_xor(s2, off, 32);
        s3 += __shfl_xor(s3, off, 32);
    }
    if (lane == 0) {
        float s[NUM_EXP] = { s0 + gb[0], s1 + gb[1], s2 + gb[2], s3 + gb[3] };
        float m = s[0];
        for (int e = 1; e < NUM_EXP; ++e) m = fmaxf(m, s[e]);
        float p[NUM_EXP], tot = 0.f;
        for (int e = 0; e < NUM_EXP; ++e) { p[e] = __expf(s[e] - m); tot += p[e]; }
        for (int e = 0; e < NUM_EXP; ++e) p[e] /= tot;
        int i0 = 0;
        for (int e = 1; e < NUM_EXP; ++e) if (p[e] > p[i0]) i0 = e;
        int i1 = -1;
        for (int e = 0; e < NUM_EXP; ++e) {
            if (e == i0) continue;
            if (i1 < 0 || p[e] > p[i1]) i1 = e;
        }
        float denom = p[i0] + p[i1] + 1e-9f;
        for (int e = 0; e < NUM_EXP; ++e) {
            float v = (e == i0 || e == i1) ? (p[e] / denom) : 0.f;
            scores[(size_t)row * NUM_EXP + e] = v;
        }
    }
}

// -------------------------------------------------------- f32 -> bf16 ------
__global__ __launch_bounds__(256)
void cvt_f32_to_bf16(const float* __restrict__ in, __bf16* __restrict__ out,
                     size_t n)
{
    size_t i = (size_t)blockIdx.x * blockDim.x + threadIdx.x;
    if (i < n) out[i] = (__bf16)in[i];
}

// ------------------------------------- W[K,N] f32 -> Wt[N,K] bf16 ----------
__global__ __launch_bounds__(256)
void transpose_to_bf16(const float* __restrict__ W, __bf16* __restrict__ Wt,
                       int K, int N)
{
    __shared__ float tile[32][33];
    int nt = blockIdx.x * 32;
    int kt = blockIdx.y * 32;
    int tx = threadIdx.x, ty = threadIdx.y;        // block (32, 8)
    for (int i = ty; i < 32; i += 8)
        tile[i][tx] = W[(size_t)(kt + i) * N + nt + tx];
    __syncthreads();
    for (int i = ty; i < 32; i += 8)
        Wt[(size_t)(nt + i) * K + kt + tx] = (__bf16)tile[tx][i];
}

// ----------------------------------------------------------- activations ---
template <int ACT>
__device__ inline float apply_act(float h)
{
    if (ACT == 0) return 0.5f * h * (1.0f + erff(h * 0.70710678118654752f)); // GELU (erf)
    if (ACT == 1) return h / (1.0f + __expf(-h));                            // SiLU
    if (ACT == 2) return fmaxf(h, 0.0f);                                     // ReLU
    if (ACT == 3) return (h >= 0.0f) ? h : 0.01f * h;                        // LeakyReLU
    return h;                                                                // none
}

// ------------------------------------------------------- bf16 WMMA GEMM ----
// C[M,N] = act(A[M,K] @ Bt[N,K]^T + bias[N]); A,Bt bf16 row-major; C f32.
// Block 256 thr = 8 waves, block tile 128(M)x256(N), wave tile 64x64.
// LDS pitch: 32 bf16 payload + 8 bf16 pad = 40 bf16 = 80 B per row.

#define LDS_PITCH 40   // bf16 elements per LDS row (80 bytes)

template <int ACT>
__global__ __launch_bounds__(256)
void gemm_bias_act(const __bf16* __restrict__ A, const __bf16* __restrict__ Bt,
                   const float* __restrict__ bias, float* __restrict__ C,
                   int M, int N, int K)
{
    __shared__ __align__(16) __bf16 As[2][128 * LDS_PITCH];   // 20 KB
    __shared__ __align__(16) __bf16 Bs[2][256 * LDS_PITCH];   // 40 KB

    const int tid   = threadIdx.x;
    const int lane  = tid & 31;
    const int wave  = tid >> 5;
    const int r     = lane & 15;
    const int ksel  = lane >> 4;          // frag chunks {ksel, ksel+2} of 4x16B
    const int tN    = N >> 8;
    const int bm    = (blockIdx.x / tN) << 7;
    const int bn    = (blockIdx.x % tN) << 8;
    const int wmOff = (wave & 1) << 6;    // 2 waves along M
    const int wnOff = (wave >> 1) << 6;   // 4 waves along N

    // ---- staging decomposition: each thread copies 16B chunks -------------
    const int srow   = tid >> 2;          // 0..63
    const int schunk = tid & 3;           // 16B chunk within 64B k-slice

    unsigned long long aBaseG = (unsigned long long)(uintptr_t)A;
    unsigned long long bBaseG = (unsigned long long)(uintptr_t)Bt;

    unsigned aOff[2], bOff[4];            // per-lane global byte offsets (k0=0)
    unsigned aLds[2][2], bLds[2][4];      // per-lane LDS byte addresses per buf
#pragma unroll
    for (int it = 0; it < 2; ++it) {
        int row = srow + it * 64;
        aOff[it] = (unsigned)(((size_t)(bm + row) * K + schunk * 8) * 2);
#pragma unroll
        for (int p = 0; p < 2; ++p)
            aLds[p][it] = (unsigned)(uintptr_t)&As[p][row * LDS_PITCH + schunk * 8];
    }
#pragma unroll
    for (int it = 0; it < 4; ++it) {
        int row = srow + it * 64;
        bOff[it] = (unsigned)(((size_t)(bn + row) * K + schunk * 8) * 2);
#pragma unroll
        for (int p = 0; p < 2; ++p)
            bLds[p][it] = (unsigned)(uintptr_t)&Bs[p][row * LDS_PITCH + schunk * 8];
    }

    // 6 async 16B copies per thread per K-step (2 for A, 4 for B)
    auto issue = [&](int p, int k0) {
        unsigned kB = (unsigned)k0 * 2;
#pragma unroll
        for (int it = 0; it < 2; ++it)
            asm volatile("global_load_async_to_lds_b128 %0, %1, %2"
                         :: "v"(aLds[p][it]), "v"(aOff[it] + kB), "s"(aBaseG)
                         : "memory");
#pragma unroll
        for (int it = 0; it < 4; ++it)
            asm volatile("global_load_async_to_lds_b128 %0, %1, %2"
                         :: "v"(bLds[p][it]), "v"(bOff[it] + kB), "s"(bBaseG)
                         : "memory");
    };

    f32x8 acc[4][4];
#pragma unroll
    for (int i = 0; i < 4; ++i)
#pragma unroll
        for (int j = 0; j < 4; ++j) acc[i][j] = (f32x8)0.0f;

    issue(0, 0);

    const int nk = K >> 5;
    for (int kk = 0; kk < nk; ++kk) {
        const int p = kk & 1;
        if (kk + 1 < nk) {
            issue(p ^ 1, (kk + 1) << 5);
            // async loads complete in order: <=6 outstanding => buf p landed
            asm volatile("s_wait_asynccnt 0x6" ::: "memory");
        } else {
            asm volatile("s_wait_asynccnt 0x0" ::: "memory");
        }
        __syncthreads();

        bf16x16 a[4];
#pragma unroll
        for (int i = 0; i < 4; ++i) {
            const __bf16* q = &As[p][(wmOff + i * 16 + r) * LDS_PITCH + ksel * 8];
            bf16x8 lo = *(const bf16x8*)q;          // K chunk ksel
            bf16x8 hi = *(const bf16x8*)(q + 16);   // K chunk ksel+2
            a[i] = __builtin_shufflevector(lo, hi, 0, 1, 2, 3, 4, 5, 6, 7,
                                                   8, 9, 10, 11, 12, 13, 14, 15);
        }
#pragma unroll
        for (int j = 0; j < 4; ++j) {
            const __bf16* q = &Bs[p][(wnOff + j * 16 + r) * LDS_PITCH + ksel * 8];
            bf16x8 lo = *(const bf16x8*)q;
            bf16x8 hi = *(const bf16x8*)(q + 16);
            bf16x16 b = __builtin_shufflevector(lo, hi, 0, 1, 2, 3, 4, 5, 6, 7,
                                                        8, 9, 10, 11, 12, 13, 14, 15);
#pragma unroll
            for (int i = 0; i < 4; ++i)
                acc[i][j] = __builtin_amdgcn_wmma_f32_16x16x32_bf16(
                    false, a[i], false, b, (short)0, acc[i][j], false, false);
        }
        __syncthreads();   // buf p may be overwritten by next iteration's issue
    }

    // Epilogue: element (q, lane) of tile (i,j):
    //   m = bm + wmOff + i*16 + ksel*8 + q,  n = bn + wnOff + j*16 + r
#pragma unroll
    for (int i = 0; i < 4; ++i) {
#pragma unroll
        for (int j = 0; j < 4; ++j) {
            int col = bn + wnOff + j * 16 + r;
            float bv = bias[col];
#pragma unroll
            for (int q = 0; q < 8; ++q) {
                int row = bm + wmOff + i * 16 + ksel * 8 + q;
                float v = acc[i][j][q] + bv;
                C[(size_t)row * N + col] = apply_act<ACT>(v);
            }
        }
    }
}

// ------------------------------------------------------- block reduction ---
static __device__ inline float block_sum(float v, float* lds)
{
    for (int off = 16; off > 0; off >>= 1) v += __shfl_xor(v, off, 32);
    int wave = threadIdx.x >> 5, lane = threadIdx.x & 31;
    if (lane == 0) lds[wave] = v;
    __syncthreads();
    if (wave == 0) {
        float t = (lane < 8) ? lds[lane] : 0.0f;
        for (int off = 4; off > 0; off >>= 1) t += __shfl_xor(t, off, 32);
        if (lane == 0) lds[0] = t;
    }
    __syncthreads();
    float r = lds[0];
    __syncthreads();
    return r;
}

// ------------------------------------ LayerNorm -> bf16 (next GEMM input) --
__global__ __launch_bounds__(256)
void ln_to_bf16(const float* __restrict__ H, const float* __restrict__ g,
                const float* __restrict__ beta, __bf16* __restrict__ O, int N)
{
    __shared__ float lds[8];
    int row = blockIdx.x;
    const float* h = H + (size_t)row * N;
    float s = 0.f, s2 = 0.f;
    for (int i = threadIdx.x; i < N; i += 256) {
        float v = h[i];
        s += v; s2 += v * v;
    }
    s  = block_sum(s, lds);
    s2 = block_sum(s2, lds);
    float invN = 1.0f / (float)N;
    float mu   = s * invN;
    float var  = fmaxf(s2 * invN - mu * mu, 0.0f);
    float inv  = rsqrtf(var + 1e-5f);
    for (int i = threadIdx.x; i < N; i += 256)
        O[(size_t)row * N + i] = (__bf16)((h[i] - mu) * inv * g[i] + beta[i]);
}

// -------------------- final LayerNorm, scale by gate score, accumulate -----
__global__ __launch_bounds__(256)
void ln_scale_accum(const float* __restrict__ H, const float* __restrict__ g,
                    const float* __restrict__ beta,
                    const float* __restrict__ scores, float* __restrict__ out,
                    int expert, int first)
{
    __shared__ float lds[8];
    const int N = INPUT_DIM;
    int row = blockIdx.x;
    const float* h = H + (size_t)row * N;
    float s = 0.f, s2 = 0.f;
    for (int i = threadIdx.x; i < N; i += 256) {
        float v = h[i];
        s += v; s2 += v * v;
    }
    s  = block_sum(s, lds);
    s2 = block_sum(s2, lds);
    float invN = 1.0f / (float)N;
    float mu   = s * invN;
    float var  = fmaxf(s2 * invN - mu * mu, 0.0f);
    float inv  = rsqrtf(var + 1e-5f);
    float w    = scores[(size_t)row * NUM_EXP + expert];
    for (int i = threadIdx.x; i < N; i += 256) {
        float v = ((h[i] - mu) * inv * g[i] + beta[i]) * w;
        size_t o = (size_t)row * N + i;
        if (first) out[o] = v;
        else       out[o] += v;
    }
}

// ------------------------------------------------------------ dispatch -----
static void launch_gemm(int act, const __bf16* A, const __bf16* Bt,
                        const float* bias, float* C, int M, int N, int K,
                        hipStream_t s)
{
    dim3 grid((M / 128) * (N / 256)), block(256);
    switch (act) {
    case 0: gemm_bias_act<0><<<grid, block, 0, s>>>(A, Bt, bias, C, M, N, K); break;
    case 1: gemm_bias_act<1><<<grid, block, 0, s>>>(A, Bt, bias, C, M, N, K); break;
    case 2: gemm_bias_act<2><<<grid, block, 0, s>>>(A, Bt, bias, C, M, N, K); break;
    case 3: gemm_bias_act<3><<<grid, block, 0, s>>>(A, Bt, bias, C, M, N, K); break;
    default: gemm_bias_act<4><<<grid, block, 0, s>>>(A, Bt, bias, C, M, N, K); break;
    }
}

extern "C" void kernel_launch(void* const* d_in, const int* in_sizes, int n_in,
                              void* d_out, int out_size, void* d_ws, size_t ws_size,
                              hipStream_t stream)
{
    (void)in_sizes; (void)n_in; (void)out_size; (void)ws_size;

    static const int DEPTH[NUM_EXP] = { 1, 2, 3, 2 };
    static const int HID[NUM_EXP]   = { 2048, 4096, 6144, 4096 };
    static const int ACT[NUM_EXP]   = { 0, 1, 2, 3 };   // gelu, silu, relu, leaky

    // ---- unpack inputs (setup_inputs dict insertion order) ----
    int idx = 0;
    const float* x = (const float*)d_in[idx++];
    const float* Wh[NUM_EXP][3]; const float* bh[NUM_EXP][3];
    const float* gh[NUM_EXP][3]; const float* bth[NUM_EXP][3];
    const float* Wo[NUM_EXP]; const float* bo[NUM_EXP];
    const float* go[NUM_EXP]; const float* bto[NUM_EXP];
    for (int e = 0; e < NUM_EXP; ++e) {
        for (int l = 0; l < DEPTH[e]; ++l) {
            Wh[e][l]  = (const float*)d_in[idx++];
            bh[e][l]  = (const float*)d_in[idx++];
            gh[e][l]  = (const float*)d_in[idx++];
            bth[e][l] = (const float*)d_in[idx++];
        }
        Wo[e]  = (const float*)d_in[idx++];
        bo[e]  = (const float*)d_in[idx++];
        go[e]  = (const float*)d_in[idx++];
        bto[e] = (const float*)d_in[idx++];
    }
    const float* gate_w = (const float*)d_in[idx++];
    const float* gate_b = (const float*)d_in[idx++];
    float* out = (float*)d_out;

    // ---- workspace carve-out (offsets 256B aligned) ----
    char* ws = (char*)d_ws;
    size_t off = 0;
    float*  scores = (float*)(ws + off);  off += (size_t)BATCH * NUM_EXP * 4;
    off = (off + 255) & ~(size_t)255;
    __bf16* xbf    = (__bf16*)(ws + off); off += (size_t)BATCH * INPUT_DIM * 2;
    off = (off + 255) & ~(size_t)255;
    float*  actF   = (float*)(ws + off);  off += (size_t)BATCH * 6144 * 4;
    off = (off + 255) & ~(size_t)255;
    __bf16* actB   = (__bf16*)(ws + off); off += (size_t)BATCH * 6144 * 2;
    off = (off + 255) & ~(size_t)255;
    __bf16* wt     = (__bf16*)(ws + off); off += (size_t)6144 * 6144 * 2;

    // ---- gating ----
    gate_kernel<<<(BATCH * 32) / 256, 256, 0, stream>>>(x, gate_w, gate_b, scores);

    // ---- x -> bf16 ----
    {
        size_t n = (size_t)BATCH * INPUT_DIM;
        cvt_f32_to_bf16<<<(unsigned)((n + 255) / 256), 256, 0, stream>>>(x, xbf, n);
    }

    // ---- experts (sequential on stream; buffers reused safely) ----
    for (int e = 0; e < NUM_EXP; ++e) {
        const __bf16* cur = xbf;
        int K = INPUT_DIM;
        int hid = HID[e];
        for (int l = 0; l < DEPTH[e]; ++l) {
            transpose_to_bf16<<<dim3(hid / 32, K / 32), dim3(32, 8), 0, stream>>>(
                Wh[e][l], wt, K, hid);
            launch_gemm(ACT[e], cur, wt, bh[e][l], actF, BATCH, hid, K, stream);
            ln_to_bf16<<<BATCH, 256, 0, stream>>>(actF, gh[e][l], bth[e][l], actB, hid);
            cur = actB;
            K = hid;
        }
        transpose_to_bf16<<<dim3(INPUT_DIM / 32, K / 32), dim3(32, 8), 0, stream>>>(
            Wo[e], wt, K, INPUT_DIM);
        launch_gemm(4, cur, wt, bo[e], actF, BATCH, INPUT_DIM, K, stream);
        ln_scale_accum<<<BATCH, 256, 0, stream>>>(actF, go[e], bto[e], scores, out,
                                                  e, e == 0 ? 1 : 0);
    }
}
